// MoETransformerBlock_23605140259538
// MI455X (gfx1250) — matrix operand order repaired and checked by reading, not compiled, not run on hardware
//
#include <hip/hip_runtime.h>

// ---------------------------------------------------------------------------
// MI455X (gfx1250) MoE transformer block, bf16 WMMA everywhere.
// 64x64 per-wave GEMM tiles (16 wmma per k-step per wave), double-buffered
// LDS with GLOBAL_LOAD_ASYNC_TO_LDS_B128 for the bf16 A operand.
// ---------------------------------------------------------------------------

typedef __attribute__((ext_vector_type(16))) __bf16 bf16x16;
typedef __attribute__((ext_vector_type(8)))  float  floatx8;
typedef int v4i_ __attribute__((__vector_size__(4 * sizeof(int))));
typedef unsigned int   u32;
typedef unsigned short u16;

#define B_  4
#define S_  2048
#define D_  1024
#define H_  16
#define DH_ 64
#define E_  8
#define DFF_ 4096
#define M_  (B_ * S_)   // 8192 tokens

// ---- bf16 helpers (RNE) ----------------------------------------------------
__device__ __forceinline__ u16 f2bf(float f) {
  u32 u = __float_as_uint(f);
  u32 r = (u + 0x7FFFu + ((u >> 16) & 1u)) >> 16;
  return (u16)r;
}
__device__ __forceinline__ float bf2f(u16 s) {
  return __uint_as_float(((u32)s) << 16);
}

union Frag { bf16x16 v; u32 u[8]; };

__device__ __forceinline__ floatx8 zerov() {
  floatx8 z;
#pragma unroll
  for (int i = 0; i < 8; ++i) z[i] = 0.0f;
  return z;
}

__device__ __forceinline__ floatx8 wmma_bf16(const bf16x16 a, const bf16x16 b, floatx8 c) {
  return __builtin_amdgcn_wmma_f32_16x16x32_bf16(false, a, false, b, (short)0, c, false, false);
}

__device__ __forceinline__ float gelu_tanh(float x) {
  const float k0 = 0.7978845608028654f, k1 = 0.044715f;
  float t = tanhf(k0 * (x + k1 * x * x * x));
  return 0.5f * x * (1.0f + t);
}

// A-operand (16x32, 16-bit) per-reg K index; B-operand (32x16, 16-bit) K index.
__device__ __forceinline__ void kidx_tables(int hl, int* kA, int* kB) {
#pragma unroll
  for (int r = 0; r < 8; ++r) {
    kA[r] = (r < 4) ? (hl * 8 + 2 * r) : (16 + hl * 8 + 2 * (r - 4));
    kB[r] = (r < 4) ? (hl * 8 + 2 * r) : (16 + hl * 8 + 2 * (r - 4));
  }
}

// ---- async global->LDS (CDNA5 GLOBAL_LOAD_ASYNC_TO_LDS_B128) ---------------
#if __has_builtin(__builtin_amdgcn_global_load_async_to_lds_b128)
#define ASYNC_LDS 1
#else
#define ASYNC_LDS 0
#endif

__device__ __forceinline__ void async_b128(void* lds, const void* g) {
#if ASYNC_LDS
  __builtin_amdgcn_global_load_async_to_lds_b128(
      (__attribute__((address_space(1))) v4i_*)(void*)g,
      (__attribute__((address_space(3))) v4i_*)lds, 0, 0);
#else
  *(uint4*)lds = *(const uint4*)g;
#endif
}

__device__ __forceinline__ void wait_async() {
#if ASYNC_LDS
#if __has_builtin(__builtin_amdgcn_s_wait_asynccnt)
  __builtin_amdgcn_s_wait_asynccnt(0);
#else
  asm volatile("s_wait_asynccnt 0" ::: "memory");
#endif
#endif
}

// ---------------------------------------------------------------------------
// Tiled GEMM: C[M,N] = A_bf16[M,K] @ W_f32[K,N] + bias, with epilogue modes.
// MODE 0: store bf16            (QKV proj)
// MODE 1: f32 out = resid + val (attention output projection, residual add)
// MODE 2: gelu, store bf16      (FFN up-proj)
// MODE 3: f32 out += val*rowscale[row]  (FFN down-proj, gated accumulate)
// Block: 256 threads (8 waves), tile 256x128, k-step 32; wave tile 64x64.
// Double-buffered LDS; A tile via async DMA, W tile via regs (f32->bf16).
// ---------------------------------------------------------------------------
#define BM 256
#define BN 128
#define BK 32
#define AST 40   // As row stride (u16): 80B rows -> 16B aligned, conflict-free
#define BST 34   // Bs row stride (u16): 17-dword bank stride, conflict-free

template <int MODE>
__global__ void __launch_bounds__(256)
gemm_bf16_kernel(const u16* __restrict__ A, const float* __restrict__ W,
                 const float* __restrict__ bias, int M, int N, int K,
                 u16* __restrict__ outB, float* __restrict__ outF,
                 const float* __restrict__ resid,
                 const float* __restrict__ rowscale, int rs_stride) {
  __shared__ u16 As[2][BM * AST];
  __shared__ u16 Bs[2][BN * BST];

  const int tid  = threadIdx.x;
  const int lane = tid & 31;
  const int wave = tid >> 5;
  const int wm   = wave & 3;       // 4 wave-rows of 64
  const int wn   = wave >> 2;      // 2 wave-cols of 64
  const int hl   = lane >> 4;
  const int l16  = lane & 15;
  const int bm   = blockIdx.y * BM;
  const int bn   = blockIdx.x * BN;

  int kA[8], kB[8];
  kidx_tables(hl, kA, kB);

  floatx8 c[4][4];
#pragma unroll
  for (int m = 0; m < 4; ++m)
#pragma unroll
    for (int n = 0; n < 4; ++n) c[m][n] = zerov();

  // Per-thread staging assignments.
  const u16*   Abase = A + (size_t)(bm + tid) * K;      // one A row per thread
  const int    wr = tid >> 3;                            // W k-row (0..31)
  const int    wc = (tid & 7) * 16;                      // W col segment
  const float* Wbase = W + (size_t)wr * N + bn + wc;
  float wreg[16];

  // ---- prologue: stage tile 0 ----
#pragma unroll
  for (int j = 0; j < 4; ++j)
    async_b128(&As[0][tid * AST + j * 8], Abase + j * 8);
#pragma unroll
  for (int j = 0; j < 16; ++j) wreg[j] = Wbase[j];
#pragma unroll
  for (int j = 0; j < 16; ++j) Bs[0][(wc + j) * BST + wr] = f2bf(wreg[j]);
  wait_async();
  __syncthreads();

  int buf = 0;
  for (int k0 = 0; k0 < K; k0 += BK) {
    const int  nb   = buf ^ 1;
    const bool more = (k0 + BK) < K;
    if (more) {
      // Issue next A tile DMA + next W tile register loads.
#pragma unroll
      for (int j = 0; j < 4; ++j)
        async_b128(&As[nb][tid * AST + j * 8], Abase + k0 + BK + j * 8);
      const float* g = Wbase + (size_t)(k0 + BK) * N;
#pragma unroll
      for (int j = 0; j < 16; ++j) wreg[j] = g[j];
    }

    // ---- compute on current buffer: 8 frag loads -> 16 wmma ----
    Frag b[4];
#pragma unroll
    for (int n = 0; n < 4; ++n) {
      const int col = wn * 64 + n * 16 + l16;
#pragma unroll
      for (int r = 0; r < 8; ++r)
        b[n].u[r] = *(const u32*)&Bs[buf][col * BST + kB[r]];
    }
#pragma unroll
    for (int m = 0; m < 4; ++m) {
      Frag a;
      const int row = wm * 64 + m * 16 + l16;
#pragma unroll
      for (int r = 0; r < 8; ++r)
        a.u[r] = *(const u32*)&As[buf][row * AST + kA[r]];
#pragma unroll
      for (int n = 0; n < 4; ++n)
        c[m][n] = wmma_bf16(a.v, b[n].v, c[m][n]);
    }

    if (more) {
#pragma unroll
      for (int j = 0; j < 16; ++j) Bs[nb][(wc + j) * BST + wr] = f2bf(wreg[j]);
      wait_async();
    }
    __syncthreads();
    buf = nb;
  }

  // ---- epilogue: C layout M = v + 8*hl, N = l16 per 16x16 frag ----
#pragma unroll
  for (int m = 0; m < 4; ++m) {
#pragma unroll
    for (int n = 0; n < 4; ++n) {
      const int col = bn + wn * 64 + n * 16 + l16;
      const float bcol = bias[col];
#pragma unroll
      for (int v = 0; v < 8; ++v) {
        const int row = bm + wm * 64 + m * 16 + v + 8 * hl;
        float val = c[m][n][v] + bcol;
        if (MODE == 2) val = gelu_tanh(val);
        const size_t idx = (size_t)row * N + col;
        if (MODE == 0 || MODE == 2) outB[idx] = f2bf(val);
        if (MODE == 1) outF[idx] = resid[idx] + val;
        if (MODE == 3) outF[idx] += val * rowscale[(size_t)row * rs_stride];
      }
    }
  }
}

// ---------------------------------------------------------------------------
// Flash attention: one WG per (q-tile of 64 rows, head, batch). 128 threads.
// ---------------------------------------------------------------------------
__global__ void __launch_bounds__(128)
attn_kernel(const u16* __restrict__ qg, const u16* __restrict__ kg,
            const u16* __restrict__ vg, u16* __restrict__ ctx) {
  __shared__ u16 Qs[64 * 64];          // [qrow][dim]
  __shared__ u16 Ks[32 * 64];          // [key][dim]
  __shared__ u16 Vs[64 * 32];          // [dim][key] (transposed)
  __shared__ u16 Ps[4 * 16 * 32];      // per-wave P tile [m][key]

  const int qt  = blockIdx.x;
  const int h   = blockIdx.y;
  const int b   = blockIdx.z;
  const int tid = threadIdx.x;
  const int lane = tid & 31;
  const int w    = tid >> 5;
  const int hl   = lane >> 4;
  const int l16  = lane & 15;
  const size_t base = (size_t)b * S_ * D_;
  const int qbase = qt * 64;

  int kA[8], kB[8];
  kidx_tables(hl, kA, kB);

  {
    const int r  = tid >> 1;
    const int c0 = (tid & 1) * 32;
    const u16* g = qg + base + (size_t)(qbase + r) * D_ + h * DH_ + c0;
#pragma unroll
    for (int j = 0; j < 16; ++j)
      *(u32*)&Qs[r * 64 + c0 + 2 * j] = ((const u32*)g)[j];
  }
  __syncthreads();

  Frag aq[2];
#pragma unroll
  for (int ks = 0; ks < 2; ++ks) {
    const int row = w * 16 + l16;
#pragma unroll
    for (int r = 0; r < 8; ++r)
      aq[ks].u[r] = *(const u32*)&Qs[row * 64 + ks * 32 + kA[r]];
  }

  floatx8 acc[4];
#pragma unroll
  for (int f = 0; f < 4; ++f) acc[f] = zerov();
  float rm[8], rl[8];
#pragma unroll
  for (int v = 0; v < 8; ++v) { rm[v] = -1e30f; rl[v] = 0.0f; }
  const float scale = 0.125f;          // 1/sqrt(64)

  for (int kb = 0; kb < S_; kb += 32) {
    __syncthreads();
    {
      const int r  = tid >> 2;
      const int c0 = (tid & 3) * 16;
      const u16* gk = kg + base + (size_t)(kb + r) * D_ + h * DH_ + c0;
#pragma unroll
      for (int j = 0; j < 8; ++j)
        *(u32*)&Ks[r * 64 + c0 + 2 * j] = ((const u32*)gk)[j];
      const u16* gv = vg + base + (size_t)(kb + r) * D_ + h * DH_ + c0;
#pragma unroll
      for (int j = 0; j < 16; ++j)
        Vs[(c0 + j) * 32 + r] = gv[j];
    }
    __syncthreads();

    floatx8 s[2];
#pragma unroll
    for (int fn = 0; fn < 2; ++fn) {
      s[fn] = zerov();
#pragma unroll
      for (int ks = 0; ks < 2; ++ks) {
        Frag bk;
        const int key = fn * 16 + l16;
#pragma unroll
        for (int r = 0; r < 8; ++r)
          bk.u[r] = *(const u32*)&Ks[key * 64 + ks * 32 + kB[r]];
        s[fn] = wmma_bf16(aq[ks].v, bk.v, s[fn]);
      }
#pragma unroll
      for (int v = 0; v < 8; ++v) s[fn][v] *= scale;
    }

    float corr[8];
#pragma unroll
    for (int v = 0; v < 8; ++v) {
      float mx = fmaxf(s[0][v], s[1][v]);
#pragma unroll
      for (int off = 8; off >= 1; off >>= 1)
        mx = fmaxf(mx, __shfl_xor(mx, off));
      const float mnew = fmaxf(rm[v], mx);
      corr[v] = __expf(rm[v] - mnew);
      rm[v] = mnew;
      const float p0 = __expf(s[0][v] - mnew);
      const float p1 = __expf(s[1][v] - mnew);
      s[0][v] = p0; s[1][v] = p1;
      float rs = p0 + p1;
#pragma unroll
      for (int off = 8; off >= 1; off >>= 1)
        rs += __shfl_xor(rs, off);
      rl[v] = rl[v] * corr[v] + rs;
    }
#pragma unroll
    for (int f = 0; f < 4; ++f)
#pragma unroll
      for (int v = 0; v < 8; ++v) acc[f][v] *= corr[v];

#pragma unroll
    for (int fn = 0; fn < 2; ++fn)
#pragma unroll
      for (int v = 0; v < 8; ++v)
        Ps[w * 512 + (v + 8 * hl) * 32 + fn * 16 + l16] = f2bf(s[fn][v]);
    Frag ap;
#pragma unroll
    for (int r = 0; r < 8; ++r)
      ap.u[r] = *(const u32*)&Ps[w * 512 + l16 * 32 + kA[r]];

#pragma unroll
    for (int fd = 0; fd < 4; ++fd) {
      Frag bv;
      const int dim = fd * 16 + l16;
#pragma unroll
      for (int r = 0; r < 8; ++r)
        bv.u[r] = *(const u32*)&Vs[dim * 32 + kB[r]];
      acc[fd] = wmma_bf16(ap.v, bv.v, acc[fd]);
    }
  }

#pragma unroll
  for (int fd = 0; fd < 4; ++fd) {
#pragma unroll
    for (int v = 0; v < 8; ++v) {
      const int row = qbase + w * 16 + v + 8 * hl;
      const int col = h * DH_ + fd * 16 + l16;
      ctx[base + (size_t)row * D_ + col] = f2bf(acc[fd][v] / rl[v]);
    }
  }
}

// ---------------------------------------------------------------------------
// LayerNorm: one row per block (256 thr, 4 cols/thr), f32 in -> bf16 out.
// ---------------------------------------------------------------------------
__global__ void __launch_bounds__(256)
ln_bf16_kernel(const float* __restrict__ x, const float* __restrict__ g,
               const float* __restrict__ be, u16* __restrict__ y) {
  const int row = blockIdx.x;
  const int tid = threadIdx.x;
  const float* xr = x + (size_t)row * D_;
  const float4 xv = ((const float4*)xr)[tid];
  float s  = xv.x + xv.y + xv.z + xv.w;
  float s2 = xv.x * xv.x + xv.y * xv.y + xv.z * xv.z + xv.w * xv.w;
#pragma unroll
  for (int off = 16; off >= 1; off >>= 1) {
    s  += __shfl_xor(s, off);
    s2 += __shfl_xor(s2, off);
  }
  __shared__ float ps[8], ps2[8];
  const int wv = tid >> 5, ln = tid & 31;
  if (ln == 0) { ps[wv] = s; ps2[wv] = s2; }
  __syncthreads();
  if (tid == 0) {
    float a = 0.f, a2 = 0.f;
#pragma unroll
    for (int i = 0; i < 8; ++i) { a += ps[i]; a2 += ps2[i]; }
    ps[0] = a; ps2[0] = a2;
  }
  __syncthreads();
  const float mu  = ps[0] * (1.0f / D_);
  const float var = ps2[0] * (1.0f / D_) - mu * mu;
  const float rs  = rsqrtf(var + 1e-5f);
  const float4 gv = ((const float4*)g)[tid];
  const float4 bv = ((const float4*)be)[tid];
  u16* yr = y + (size_t)row * D_ + tid * 4;
  yr[0] = f2bf((xv.x - mu) * rs * gv.x + bv.x);
  yr[1] = f2bf((xv.y - mu) * rs * gv.y + bv.y);
  yr[2] = f2bf((xv.z - mu) * rs * gv.z + bv.z);
  yr[3] = f2bf((xv.w - mu) * rs * gv.w + bv.w);
}

// ---------------------------------------------------------------------------
// Gating: probs = softmax(e_in @ Wg); top-2 renormalized -> gw[token][E_].
// ---------------------------------------------------------------------------
__global__ void __launch_bounds__(256)
gate_kernel(const u16* __restrict__ e_in, const float* __restrict__ Wg,
            float* __restrict__ gw) {
  __shared__ float W[D_ * E_];
  const int tid = threadIdx.x;
  for (int i = tid; i < D_ * E_; i += 256) W[i] = Wg[i];
  __syncthreads();

  const int t = blockIdx.x * 256 + tid;
  const u16* xr = e_in + (size_t)t * D_;
  float acc[E_];
#pragma unroll
  for (int e = 0; e < E_; ++e) acc[e] = 0.0f;
  for (int kk = 0; kk < D_; ++kk) {
    const float xv = bf2f(xr[kk]);
#pragma unroll
    for (int e = 0; e < E_; ++e) acc[e] += xv * W[kk * E_ + e];
  }
  float mx = acc[0];
#pragma unroll
  for (int e = 1; e < E_; ++e) mx = fmaxf(mx, acc[e]);
  float sum = 0.0f;
#pragma unroll
  for (int e = 0; e < E_; ++e) { acc[e] = __expf(acc[e] - mx); sum += acc[e]; }
  const float inv = 1.0f / sum;
#pragma unroll
  for (int e = 0; e < E_; ++e) acc[e] *= inv;
  int i1 = 0;
#pragma unroll
  for (int e = 1; e < E_; ++e) if (acc[e] > acc[i1]) i1 = e;
  int i2 = (i1 == 0) ? 1 : 0;
#pragma unroll
  for (int e = 0; e < E_; ++e) if (e != i1 && acc[e] > acc[i2]) i2 = e;
  const float tot = acc[i1] + acc[i2];
#pragma unroll
  for (int e = 0; e < E_; ++e)
    gw[(size_t)t * E_ + e] = (e == i1) ? acc[i1] / tot : (e == i2) ? acc[i2] / tot : 0.0f;
}

// ---------------------------------------------------------------------------
extern "C" void kernel_launch(void* const* d_in, const int* in_sizes, int n_in,
                              void* d_out, int out_size, void* d_ws, size_t ws_size,
                              hipStream_t stream) {
  const float* x   = (const float*)d_in[0];
  const float* Wq  = (const float*)d_in[1];
  const float* bq  = (const float*)d_in[2];
  const float* Wk  = (const float*)d_in[3];
  const float* bk  = (const float*)d_in[4];
  const float* Wv  = (const float*)d_in[5];
  const float* bv  = (const float*)d_in[6];
  const float* Wo  = (const float*)d_in[7];
  const float* bo  = (const float*)d_in[8];
  const float* g1  = (const float*)d_in[9];
  const float* be1 = (const float*)d_in[10];
  const float* g2  = (const float*)d_in[11];
  const float* be2 = (const float*)d_in[12];
  const float* Wg  = (const float*)d_in[13];
  const float* W1  = (const float*)d_in[14];
  const float* B1  = (const float*)d_in[15];
  const float* W2  = (const float*)d_in[16];
  const float* B2  = (const float*)d_in[17];
  float* out = (float*)d_out;

  // Workspace carve-out (~161 MB).
  char* ws = (char*)d_ws;
  size_t off = 0;
  auto carve = [&](size_t bytes) -> char* {
    char* p = ws + off;
    off += (bytes + 255) & ~(size_t)255;
    return p;
  };
  u16*   a_in = (u16*)carve((size_t)M_ * D_ * 2);
  u16*   qb   = (u16*)carve((size_t)M_ * D_ * 2);
  u16*   kb   = (u16*)carve((size_t)M_ * D_ * 2);
  u16*   vb   = (u16*)carve((size_t)M_ * D_ * 2);
  u16*   ctxb = (u16*)carve((size_t)M_ * D_ * 2);
  u16*   e_in = (u16*)carve((size_t)M_ * D_ * 2);
  u16*   hb   = (u16*)carve((size_t)M_ * DFF_ * 2);
  float* gw   = (float*)carve((size_t)M_ * E_ * 4);
  (void)ws_size; (void)in_sizes; (void)n_in; (void)out_size;

  const dim3 blk256(256), blk128(128);
  const dim3 gProj(D_ / BN, M_ / BM);     // (8, 32)
  const dim3 gFF1(DFF_ / BN, M_ / BM);    // (32, 32)

  // --- attention sublayer ---
  ln_bf16_kernel<<<M_, blk256, 0, stream>>>(x, g1, be1, a_in);
  gemm_bf16_kernel<0><<<gProj, blk256, 0, stream>>>(a_in, Wq, bq, M_, D_, D_, qb, nullptr, nullptr, nullptr, 0);
  gemm_bf16_kernel<0><<<gProj, blk256, 0, stream>>>(a_in, Wk, bk, M_, D_, D_, kb, nullptr, nullptr, nullptr, 0);
  gemm_bf16_kernel<0><<<gProj, blk256, 0, stream>>>(a_in, Wv, bv, M_, D_, D_, vb, nullptr, nullptr, nullptr, 0);
  attn_kernel<<<dim3(S_ / 64, H_, B_), blk128, 0, stream>>>(qb, kb, vb, ctxb);
  gemm_bf16_kernel<1><<<gProj, blk256, 0, stream>>>(ctxb, Wo, bo, M_, D_, D_, nullptr, out, x, nullptr, 0);

  // --- MoE FFN sublayer ---
  ln_bf16_kernel<<<M_, blk256, 0, stream>>>(out, g2, be2, e_in);
  gate_kernel<<<M_ / 256, blk256, 0, stream>>>(e_in, Wg, gw);
  for (int e = 0; e < E_; ++e) {
    const float* W1e = W1 + (size_t)e * D_ * DFF_;
    const float* B1e = B1 + (size_t)e * DFF_;
    const float* W2e = W2 + (size_t)e * DFF_ * D_;
    const float* B2e = B2 + (size_t)e * D_;
    gemm_bf16_kernel<2><<<gFF1, blk256, 0, stream>>>(e_in, W1e, B1e, M_, DFF_, D_, hb, nullptr, nullptr, nullptr, 0);
    gemm_bf16_kernel<3><<<gProj, blk256, 0, stream>>>(hb, W2e, B2e, M_, D_, DFF_, nullptr, out, nullptr, gw + e, E_);
  }
}
